// Router_40389872451653
// MI455X (gfx1250) — compile-verified, matrix operand order checked by reading, not run on hardware
//
#include <hip/hip_runtime.h>
#include <hip/hip_bf16.h>

// Static problem config (matches reference)
#define NTOK     4096      // B*T
#define DDIM     1024
#define NEXP     8
#define TOPK     2
#define EXP_CAP  1280

typedef __attribute__((ext_vector_type(16))) _Float16 v16h;
typedef __attribute__((ext_vector_type(8)))  float    v8f;
typedef __attribute__((ext_vector_type(4)))  float    f32x4;

// ---------------------------------------------------------------------------
// Kernel 1: streaming zero-fill of cb_weight + sec_mask (336 MB of zeros).
// Non-temporal 128-bit stores: write-once traffic, keep it out of L2.
// This is the roofline-dominant cost: 336 MB / 23.3 TB/s ~= 14.4 us.
// ---------------------------------------------------------------------------
__global__ __launch_bounds__(256) void fill_zero_k(f32x4* __restrict__ p, size_t n4) {
  size_t i      = (size_t)blockIdx.x * blockDim.x + threadIdx.x;
  size_t stride = (size_t)gridDim.x * blockDim.x;
  f32x4 z = {0.f, 0.f, 0.f, 0.f};
  for (; i < n4; i += stride) {
    __builtin_nontemporal_store(z, p + i);
  }
}

// ---------------------------------------------------------------------------
// Kernel 2: router logits via WMMA. One wave computes a 16-token x 16-col
// tile. Columns 0..7 = experts; columns 8..15 are pointed at experts 0..7
// again (duplicate results, never written) so the inner loop is completely
// branch-free and every lane does identical work -> no exec toggling, no
// B-fragment zeroing, WMMA runs with EXEC all-1s as required.
//
// Fragment layouts per ISA 7.12.2 (16-bit 16x32):
//   A: lane l holds row M = l&15; halves h=0..7 -> K=(l>=16?8:0)+h,
//      halves h=8..15 -> K=16+(l>=16?8:0)+(h-8)
//   B: symmetric, lane l holds column N = l&15, same K map.
//   C/D (32-bit 16x16): lane l = column N=l&15; VGPR r -> row r+(l>=16?8:0).
// ---------------------------------------------------------------------------
__global__ __launch_bounds__(128) void router_logits_wmma_k(
    const float* __restrict__ x,   // (NTOK, DDIM)
    const float* __restrict__ w,   // (NEXP, DDIM)
    float* __restrict__ logits)    // (NTOK, NEXP)
{
  const int lane = threadIdx.x & 31;
  const int wave = threadIdx.x >> 5;
  const int tile = blockIdx.x * 4 + wave;   // 16-token tile index, 256 tiles
  const int row0 = tile * 16;

  const int half = lane >> 4;   // 0: K-base +0, 1: K-base +8
  const int mn   = lane & 15;   // A: token row in tile; B: column

  const float* xrow = x + (size_t)(row0 + mn) * DDIM + half * 8;
  const float* wrow = w + (size_t)(mn & (NEXP - 1)) * DDIM + half * 8;

  v8f acc = {};
  #pragma unroll 2
  for (int k0 = 0; k0 < DDIM; k0 += 32) {
    const f32x4* xa = (const f32x4*)(xrow + k0);
    const f32x4* xb = (const f32x4*)(xrow + k0 + 16);
    const f32x4* wa = (const f32x4*)(wrow + k0);
    const f32x4* wb = (const f32x4*)(wrow + k0 + 16);
    f32x4 a0 = xa[0], a1 = xa[1], a2 = xb[0], a3 = xb[1];
    f32x4 b0 = wa[0], b1 = wa[1], b2 = wb[0], b3 = wb[1];

    v16h a, b;
    #pragma unroll
    for (int j = 0; j < 4; ++j) {
      a[j]      = (_Float16)a0[j];
      a[4 + j]  = (_Float16)a1[j];
      a[8 + j]  = (_Float16)a2[j];
      a[12 + j] = (_Float16)a3[j];
      b[j]      = (_Float16)b0[j];
      b[4 + j]  = (_Float16)b1[j];
      b[8 + j]  = (_Float16)b2[j];
      b[12 + j] = (_Float16)b3[j];
    }
    acc = __builtin_amdgcn_wmma_f32_16x16x32_f16(
        /*neg_a=*/false, a, /*neg_b=*/false, b,
        /*c_mod=*/(short)0, acc, /*reuse_a=*/false, /*reuse_b=*/false);
  }

  // Write out: lane holds column N=mn; VGPR r -> token row r + half*8.
  // Columns 8..15 are duplicates -> skip.
  if (mn < NEXP) {
    #pragma unroll
    for (int r = 0; r < 8; ++r) {
      int token = row0 + r + half * 8;
      logits[(size_t)token * NEXP + mn] = acc[r];
    }
  }
}

// ---------------------------------------------------------------------------
// Kernel 3: per-token top-2 + softmax over the two selected logits.
// Tie-break: first (lowest) index wins, matching jax.lax.top_k.
// ---------------------------------------------------------------------------
__global__ __launch_bounds__(256) void topk_softmax_k(
    const float* __restrict__ logits,   // (NTOK, NEXP)
    int*   __restrict__ sel_idx,        // (TOPK, NTOK) k-major
    float* __restrict__ sel_w)          // (TOPK, NTOK)
{
  int t = blockIdx.x * blockDim.x + threadIdx.x;
  if (t >= NTOK) return;

  float l[NEXP];
  #pragma unroll
  for (int e = 0; e < NEXP; ++e) l[e] = logits[(size_t)t * NEXP + e];

  int i0 = 0;
  #pragma unroll
  for (int e = 1; e < NEXP; ++e) if (l[e] > l[i0]) i0 = e;

  int i1 = (i0 == 0) ? 1 : 0;
  #pragma unroll
  for (int e = 0; e < NEXP; ++e) {
    if (e == i0) continue;
    if (l[e] > l[i1]) i1 = e;
  }

  // softmax over {l[i0], l[i1]} with max subtracted (l[i0] >= l[i1])
  float s  = expf(l[i1] - l[i0]);
  float p0 = 1.0f / (1.0f + s);
  float p1 = s * p0;

  sel_idx[0 * NTOK + t] = i0;
  sel_idx[1 * NTOK + t] = i1;
  sel_w[0 * NTOK + t]   = p0;
  sel_w[1 * NTOK + t]   = p1;
}

// ---------------------------------------------------------------------------
// Kernel 4: capacity scan + scatter. One block, 8 wave32s, one wave per
// expert. Scans the 8192 flat (k*NTOK + t) entries in k-major order in 256
// chunks of 32; wave prefix via ballot_w32 + popcount reproduces the exact
// sequential cumsum semantics of the reference. Kept entries scatter the
// router prob into cb_weight[t][e][rank] and 1.0 into sec_mask.
// ---------------------------------------------------------------------------
__global__ __launch_bounds__(256) void scan_scatter_k(
    const int*   __restrict__ sel_idx,   // (TOPK, NTOK)
    const float* __restrict__ sel_w,     // (TOPK, NTOK)
    float* __restrict__ used_cap,        // (NEXP)
    float* __restrict__ cb_weight,       // (NTOK, NEXP, EXP_CAP)
    float* __restrict__ sec_mask)        // (NTOK, NEXP, EXP_CAP)
{
  const int lane = threadIdx.x & 31;
  const int e    = threadIdx.x >> 5;     // expert id, 0..7

  int base = 0;
  const int nflat = TOPK * NTOK;         // 8192
  for (int c = 0; c < nflat; c += 32) {
    int idx   = c + lane;                // flat = k*NTOK + t
    int exp   = sel_idx[idx];
    bool hit  = (exp == e);
    unsigned bal   = __builtin_amdgcn_ballot_w32(hit);
    int      prior = __builtin_popcount(bal & ((1u << lane) - 1u));
    int      rank  = base + prior;
    if (hit && rank < EXP_CAP) {
      int t = idx & (NTOK - 1);
      size_t off = ((size_t)t * NEXP + e) * EXP_CAP + rank;
      float wgt = sel_w[idx];
      cb_weight[off] = wgt;
      sec_mask[off]  = 1.0f;
    }
    base += __builtin_popcount(bal);
  }
  if (lane == 0) {
    int uc = base < EXP_CAP ? base : EXP_CAP;
    used_cap[e] = (float)uc;
  }
}

// ---------------------------------------------------------------------------
extern "C" void kernel_launch(void* const* d_in, const int* in_sizes, int n_in,
                              void* d_out, int out_size, void* d_ws, size_t ws_size,
                              hipStream_t stream) {
  (void)in_sizes; (void)n_in; (void)out_size; (void)ws_size;

  const float* x = (const float*)d_in[0];   // (B,T,D) = (NTOK, DDIM)
  const float* w = (const float*)d_in[1];   // (NEXP, DDIM)

  float* out      = (float*)d_out;
  float* used_cap = out;                                         // 8 elems
  float* cb       = out + NEXP;                                  // NTOK*NEXP*EXP_CAP
  float* sm       = cb + (size_t)NTOK * NEXP * EXP_CAP;          // same size

  // Workspace layout
  float* logits = (float*)d_ws;                                  // NTOK*NEXP f32
  int*   sidx   = (int*)(logits + (size_t)NTOK * NEXP);          // TOPK*NTOK i32
  float* sw     = (float*)(sidx + (size_t)TOPK * NTOK);          // TOPK*NTOK f32

  // 1) zero-fill cb_weight + sec_mask (contiguous): 2*NTOK*NEXP*EXP_CAP floats
  size_t nfill4 = ((size_t)2 * NTOK * NEXP * EXP_CAP) / 4;       // 20,971,520 float4
  fill_zero_k<<<2048, 256, 0, stream>>>((f32x4*)cb, nfill4);

  // 2) logits GEMM via WMMA: 256 tiles of 16 tokens, 4 waves per block
  router_logits_wmma_k<<<NTOK / 16 / 4, 128, 0, stream>>>(x, w, logits);

  // 3) top-2 + softmax
  topk_softmax_k<<<NTOK / 256, 256, 0, stream>>>(logits, sidx, sw);

  // 4) capacity scan + scatter + used_cap
  scan_scatter_k<<<1, 256, 0, stream>>>(sidx, sw, used_cap, cb, sm);
}
